// HybridEncoderLayer_34265249088189
// MI455X (gfx1250) — matrix operand order repaired and checked by reading, not compiled, not run on hardware
//
#include <hip/hip_runtime.h>
#include <hip/hip_bf16.h>

// ---------------------------------------------------------------------------
// HybridEncoderLayer (Mamba mixer + FFN) for gfx1250 (MI455X, wave32, WMMA)
//
// GEMMs: v_wmma_f32_16x16x32_f16, 128-thread blocks, 128x64 tile, BK=32,
// double-buffered LDS, async global->LDS (ASYNCcnt) when the toolchain
// declares the gfx1250 builtins, 8 WMMAs per wave per K-step.
// Selective scan: fp32 VALU recurrence, 16 lanes per (batch,channel).
// ---------------------------------------------------------------------------

typedef _Float16 v16h __attribute__((ext_vector_type(16)));
typedef float    v8f  __attribute__((ext_vector_type(8)));

#if __has_builtin(__builtin_amdgcn_global_load_async_to_lds_b128) && \
    __has_builtin(__builtin_amdgcn_s_wait_asynccnt)
#define USE_ASYNC_LDS 1
// ROCm 7.2 clang-22 prototype takes pointers to 16-byte int vectors.
typedef int v4i __attribute__((vector_size(16)));
#endif

#define NTOK   4096   // 4 batches * 1024 tokens
#define DMODEL 768
#define DINNER 1536
#define DXR    3072   // 2*DINNER
#define DPROJ  80     // dt_rank + 2*n
#define DTRANK 48
#define NSTATE 16
#define DFF    3072
#define SEQL   1024
#define NBATCH 4

#define BM 128        // block tile M (4 waves x 32 rows)
#define BN 64         // block tile N
#define BK 32         // K step
#define LDT 40        // padded LDS row stride (halves) -> 80B rows, 16B aligned

// ---------------------------------------------------------------------------
// Stage one (BM x BK) A-tile and (BN x BK) B-tile into LDS.
// Full interior chunks go through async global->LDS b128 when available.
// Remainder chunks (K or N edge) take a guarded scalar path.
// ---------------------------------------------------------------------------
__device__ __forceinline__
void stage_tile(const _Float16* __restrict__ A, const _Float16* __restrict__ B,
                int N, int K, int m0, int n0, int k0,
                _Float16* sA, _Float16* sB, int tid)
{
    const bool kfull = (k0 + BK) <= K;
    const bool nfull = (n0 + BN) <= N;

    // A tile: BM x BK = 4096 halves = 512 chunks of 8; 4 chunks per thread.
    #pragma unroll
    for (int j = 0; j < 4; ++j) {
        const int chunk = tid + j * 128;
        const int r  = chunk >> 2;          // tile row 0..127 (always valid, M%128==0)
        const int cc = (chunk & 3) << 3;    // 0,8,16,24
        const _Float16* gp = A + (size_t)(m0 + r) * K + k0 + cc;
        _Float16* lp = &sA[r * LDT + cc];
        if (kfull) {
#ifdef USE_ASYNC_LDS
            __builtin_amdgcn_global_load_async_to_lds_b128((v4i*)gp, (v4i*)lp, 0, 0);
#else
            *(uint4*)lp = *(const uint4*)gp;
            __builtin_prefetch(gp + BK, 0, 1);   // global_prefetch_b8
#endif
        } else {
            #pragma unroll
            for (int e = 0; e < 8; ++e) {
                int k = k0 + cc + e;
                lp[e] = (k < K) ? A[(size_t)(m0 + r) * K + k] : (_Float16)0.0f;
            }
        }
    }
    // B tile: BN x BK = 2048 halves = 256 chunks of 8; 2 chunks per thread.
    #pragma unroll
    for (int j = 0; j < 2; ++j) {
        const int chunk = tid + j * 128;
        const int r  = chunk >> 2;          // tile row 0..63
        const int cc = (chunk & 3) << 3;
        const _Float16* gp = B + (size_t)(n0 + r) * K + k0 + cc;
        _Float16* lp = &sB[r * LDT + cc];
        if (kfull && nfull) {
#ifdef USE_ASYNC_LDS
            __builtin_amdgcn_global_load_async_to_lds_b128((v4i*)gp, (v4i*)lp, 0, 0);
#else
            *(uint4*)lp = *(const uint4*)gp;
#endif
        } else {
            const bool rowok = (n0 + r) < N;
            #pragma unroll
            for (int e = 0; e < 8; ++e) {
                int k = k0 + cc + e;
                lp[e] = (rowok && k < K) ? B[(size_t)(n0 + r) * K + k] : (_Float16)0.0f;
            }
        }
    }
}

// ---------------------------------------------------------------------------
// Generic f16 WMMA GEMM:  C[M,N] = A[M,K] * B[N,K]^T  (+bias +addsrc, epilogue)
// mode 0: Cf = acc (+bias) (+addsrc)       [fp32 store]
// mode 1: Ch = f16(gelu_exact(acc + bias)) [f16 store]
// Requires M % 128 == 0, K % 8 == 0, K >= 32 (true for all calls here).
// ---------------------------------------------------------------------------
__global__ __launch_bounds__(128)
void gemm_f16_wmma(const _Float16* __restrict__ A, const _Float16* __restrict__ B,
                   int M, int N, int K,
                   const float* __restrict__ bias, const float* __restrict__ addsrc,
                   int mode, float* __restrict__ Cf, _Float16* __restrict__ Ch)
{
    __shared__ __align__(16) _Float16 sAbuf[2][BM * LDT];
    __shared__ __align__(16) _Float16 sBbuf[2][BN * LDT];

    const int tid  = threadIdx.x;
    const int wave = tid >> 5;
    const int lane = tid & 31;
    const int hh   = lane >> 4;   // lane-half selector (ISA 7.12.2 layout)
    const int l16  = lane & 15;
    const int m0   = blockIdx.y * BM;
    const int n0   = blockIdx.x * BN;

    v8f acc[2][4];
    #pragma unroll
    for (int mi = 0; mi < 2; ++mi)
        #pragma unroll
        for (int s = 0; s < 4; ++s)
            #pragma unroll
            for (int r = 0; r < 8; ++r) acc[mi][s][r] = 0.0f;

    // prologue: stage tile 0 into buffer 0
    stage_tile(A, B, N, K, m0, n0, 0, sAbuf[0], sBbuf[0], tid);
#ifdef USE_ASYNC_LDS
    __builtin_amdgcn_s_wait_asynccnt(0);
#endif
    __syncthreads();

    int p = 0;
    for (int k0 = 0; k0 < K; k0 += BK) {
        // stage next tile into the other buffer while computing this one
        if (k0 + BK < K)
            stage_tile(A, B, N, K, m0, n0, k0 + BK, sAbuf[p ^ 1], sBbuf[p ^ 1], tid);

        // ---- fragments per ISA 7.12.2 lane layout; 8 WMMAs per wave ----
        const _Float16* bA = &sAbuf[p][0];
        const _Float16* bB = &sBbuf[p][0];
        v16h af[2];
        #pragma unroll
        for (int mi = 0; mi < 2; ++mi) {
            // A frag: row M = wave*32 + mi*16 + l16
            // element j -> K = (j<8 ? hh*8+j : 16+hh*8+(j-8))
            const _Float16* ar = &bA[(wave * 32 + mi * 16 + l16) * LDT];
            #pragma unroll
            for (int j = 0; j < 8; ++j) {
                af[mi][j]     = ar[hh * 8 + j];
                af[mi][8 + j] = ar[16 + hh * 8 + j];
            }
        }
        #pragma unroll
        for (int s = 0; s < 4; ++s) {
            // B frag: col N = s*16 + l16 ; element j -> K = (j<8 ? hh*8+j : 16+hh*8+(j-8))
            const _Float16* br = &bB[(s * 16 + l16) * LDT];
            v16h bf;
            #pragma unroll
            for (int j = 0; j < 8; ++j) { bf[j] = br[hh * 8 + j]; bf[8 + j] = br[16 + hh * 8 + j]; }
            acc[0][s] = __builtin_amdgcn_wmma_f32_16x16x32_f16(
                false, af[0], false, bf, (short)0, acc[0][s], false, false);
            acc[1][s] = __builtin_amdgcn_wmma_f32_16x16x32_f16(
                false, af[1], false, bf, (short)0, acc[1][s], false, false);
        }

#ifdef USE_ASYNC_LDS
        __builtin_amdgcn_s_wait_asynccnt(0);
#endif
        __syncthreads();
        p ^= 1;
    }

    // ---- epilogue: C layout lanes<16 -> M=r, lanes>=16 -> M=8+r ----
    #pragma unroll
    for (int mi = 0; mi < 2; ++mi) {
        #pragma unroll
        for (int s = 0; s < 4; ++s) {
            const int col = n0 + s * 16 + l16;
            if (col >= N) continue;
            #pragma unroll
            for (int r = 0; r < 8; ++r) {
                const int row = m0 + wave * 32 + mi * 16 + hh * 8 + r;
                float v = acc[mi][s][r];
                if (bias)   v += bias[col];
                if (addsrc) v += addsrc[(size_t)row * N + col];
                if (mode == 1) {
                    float gv = 0.5f * v * (1.0f + erff(v * 0.70710678118654752f));
                    Ch[(size_t)row * N + col] = (_Float16)gv;
                } else {
                    Cf[(size_t)row * N + col] = v;
                }
            }
        }
    }
}

// ---------------------------------------------------------------------------
// LayerNorm (fp32 in) -> f16 out. One block per token.
// ---------------------------------------------------------------------------
__global__ __launch_bounds__(256)
void ln_to_f16(const float* __restrict__ x, const float* __restrict__ g,
               const float* __restrict__ b, _Float16* __restrict__ out, int D)
{
    __shared__ float sred[256];
    const int t = blockIdx.x;
    const float* xr = x + (size_t)t * D;

    float lsum = 0.f;
    for (int i = threadIdx.x; i < D; i += 256) lsum += xr[i];
    sred[threadIdx.x] = lsum; __syncthreads();
    for (int off = 128; off > 0; off >>= 1) {
        if (threadIdx.x < off) sred[threadIdx.x] += sred[threadIdx.x + off];
        __syncthreads();
    }
    const float mu = sred[0] / (float)D;
    __syncthreads();

    float lvar = 0.f;
    for (int i = threadIdx.x; i < D; i += 256) {
        float dv = xr[i] - mu; lvar += dv * dv;
    }
    sred[threadIdx.x] = lvar; __syncthreads();
    for (int off = 128; off > 0; off >>= 1) {
        if (threadIdx.x < off) sred[threadIdx.x] += sred[threadIdx.x + off];
        __syncthreads();
    }
    const float rstd = rsqrtf(sred[0] / (float)D + 1e-5f);
    for (int i = threadIdx.x; i < D; i += 256)
        out[(size_t)t * D + i] = (_Float16)((xr[i] - mu) * rstd * g[i] + b[i]);
}

// ---------------------------------------------------------------------------
// fp32 -> f16 weight conversion
// ---------------------------------------------------------------------------
__global__ void cvt_f32_to_f16(const float* __restrict__ src,
                               _Float16* __restrict__ dst, int n)
{
    int i = blockIdx.x * 256 + threadIdx.x;
    if (i < n) dst[i] = (_Float16)src[i];
}

// ---------------------------------------------------------------------------
// Depthwise causal conv (d_conv=4) + SiLU on the xs half of x_and_res.
// ---------------------------------------------------------------------------
__global__ void conv_silu(const float* __restrict__ xres,
                          const float* __restrict__ cw, const float* __restrict__ cb,
                          float* __restrict__ xs_f32, _Float16* __restrict__ xs_f16)
{
    const int i = blockIdx.x * 256 + threadIdx.x;   // over NTOK*DINNER
    if (i >= NTOK * DINNER) return;
    const int t = i / DINNER, d = i - t * DINNER;
    const int b = t >> 10, l = t & 1023;
    float acc = cb[d];
    #pragma unroll
    for (int k = 0; k < 4; ++k) {
        const int ls = l - 3 + k;
        if (ls >= 0)
            acc += cw[d * 4 + k] * xres[((size_t)(b << 10) + ls) * DXR + d];
    }
    const float s = acc / (1.f + expf(-acc));
    xs_f32[i] = s;
    xs_f16[i] = (_Float16)s;
}

// ---------------------------------------------------------------------------
// Extract dt columns (proj[:, :48]) -> f16 for the dt_proj GEMM.
// ---------------------------------------------------------------------------
__global__ void extract_dt(const float* __restrict__ proj,
                           _Float16* __restrict__ dt)
{
    const int i = blockIdx.x * 256 + threadIdx.x;
    if (i >= NTOK * DTRANK) return;
    const int t = i / DTRANK, r = i - t * DTRANK;
    dt[i] = (_Float16)proj[(size_t)t * DPROJ + r];
}

// ---------------------------------------------------------------------------
// Selective scan: 16 lanes per (batch, channel); lane = SSM state index.
// Serial over L=1024, shfl_xor tree reduction over the 16 states each step.
// ---------------------------------------------------------------------------
__global__ __launch_bounds__(256)
void selective_scan_kernel(const float* __restrict__ draw,   // (NTOK, DINNER) pre-softplus
                           const float* __restrict__ dtb,    // (DINNER)
                           const float* __restrict__ A_log,  // (DINNER, NSTATE)
                           const float* __restrict__ Dp,     // (DINNER)
                           const float* __restrict__ xs,     // (NTOK, DINNER)
                           const float* __restrict__ proj,   // (NTOK, DPROJ)
                           float* __restrict__ y)            // (NTOK, DINNER)
{
    const int tid = threadIdx.x;
    const int ch  = blockIdx.x * 16 + (tid >> 4);   // 0 .. NBATCH*DINNER-1
    const int n   = tid & 15;
    const int b   = ch / DINNER;
    const int d   = ch - b * DINNER;

    const float Av   = -expf(A_log[d * NSTATE + n]);
    const float Dv   = Dp[d];
    const float bias = dtb[d];
    float h = 0.f;
    const size_t base = (size_t)b * SEQL;

    for (int l = 0; l < SEQL; ++l) {
        const size_t t = base + l;
        const float dr  = draw[t * DINNER + d] + bias;
        const float dpl = (dr > 20.f) ? dr : log1pf(expf(dr));   // softplus
        const float u   = xs[t * DINNER + d];
        const float Bm  = proj[t * DPROJ + DTRANK + n];
        const float Cm  = proj[t * DPROJ + DTRANK + NSTATE + n];
        h = expf(dpl * Av) * h + (dpl * u) * Bm;
        float c = h * Cm;
        c += __shfl_xor(c, 8, 32);
        c += __shfl_xor(c, 4, 32);
        c += __shfl_xor(c, 2, 32);
        c += __shfl_xor(c, 1, 32);
        if (n == 0) y[t * DINNER + d] = c + u * Dv;
    }
}

// ---------------------------------------------------------------------------
// y * silu(res) -> f16 (input to out_proj GEMM)
// ---------------------------------------------------------------------------
__global__ void gate_silu(const float* __restrict__ y, const float* __restrict__ xres,
                          _Float16* __restrict__ out)
{
    const int i = blockIdx.x * 256 + threadIdx.x;
    if (i >= NTOK * DINNER) return;
    const int t = i / DINNER, d = i - t * DINNER;
    const float r = xres[(size_t)t * DXR + DINNER + d];
    const float g = r / (1.f + expf(-r));
    out[i] = (_Float16)(y[i] * g);
}

// ---------------------------------------------------------------------------
// Host-side orchestration
// ---------------------------------------------------------------------------
extern "C" void kernel_launch(void* const* d_in, const int* in_sizes, int n_in,
                              void* d_out, int out_size, void* d_ws, size_t ws_size,
                              hipStream_t stream)
{
    (void)in_sizes; (void)n_in; (void)out_size; (void)ws_size;

    const float* src   = (const float*)d_in[0];
    const float* n1g   = (const float*)d_in[1];
    const float* n1b   = (const float*)d_in[2];
    const float* w_inp = (const float*)d_in[3];
    const float* convw = (const float*)d_in[4];
    const float* convb = (const float*)d_in[5];
    const float* w_xp  = (const float*)d_in[6];
    const float* w_dt  = (const float*)d_in[7];
    const float* dt_b  = (const float*)d_in[8];
    const float* A_log = (const float*)d_in[9];
    const float* D_p   = (const float*)d_in[10];
    const float* w_out = (const float*)d_in[11];
    const float* n2g   = (const float*)d_in[12];
    const float* n2b   = (const float*)d_in[13];
    const float* w1    = (const float*)d_in[14];
    const float* b1    = (const float*)d_in[15];
    const float* w2    = (const float*)d_in[16];
    const float* b2    = (const float*)d_in[17];

    // ---- carve workspace (deterministic offsets, 256B aligned) ----
    char* wp = (char*)d_ws;
    auto carve = [&](size_t bytes) -> char* {
        char* p = wp; wp += (bytes + 255) & ~(size_t)255; return p;
    };
    _Float16* f_win  = (_Float16*)carve((size_t)DXR * DMODEL * 2);     // in_proj  f16
    _Float16* f_wxp  = (_Float16*)carve((size_t)DPROJ * DINNER * 2);   // x_proj   f16
    _Float16* f_wdt  = (_Float16*)carve((size_t)DINNER * DTRANK * 2);  // dt_proj  f16
    _Float16* f_wout = (_Float16*)carve((size_t)DMODEL * DINNER * 2);  // out_proj f16
    _Float16* f_w1   = (_Float16*)carve((size_t)DFF * DMODEL * 2);     // ffn w1   f16
    _Float16* f_w2   = (_Float16*)carve((size_t)DMODEL * DFF * 2);     // ffn w2   f16
    _Float16* h1_16  = (_Float16*)carve((size_t)NTOK * DMODEL * 2);    // LN1 out
    float*    xres   = (float*)   carve((size_t)NTOK * DXR * 4);       // in_proj out
    float*    xs32   = (float*)   carve((size_t)NTOK * DINNER * 4);    // conv+silu
    _Float16* xs16   = (_Float16*)carve((size_t)NTOK * DINNER * 2);
    float*    projo  = (float*)   carve((size_t)NTOK * DPROJ * 4);     // x_proj out
    _Float16* dt16   = (_Float16*)carve((size_t)NTOK * DTRANK * 2);
    float*    draw   = (float*)   carve((size_t)NTOK * DINNER * 4);    // dt_proj out
    float*    yssm   = (float*)   carve((size_t)NTOK * DINNER * 4);    // scan out
    _Float16* yg16   = (_Float16*)carve((size_t)NTOK * DINNER * 2);    // gated
    float*    x1     = (float*)   carve((size_t)NTOK * DMODEL * 4);    // src + mixer
    _Float16* h2_16  = (_Float16*)carve((size_t)NTOK * DMODEL * 2);    // LN2 out
    _Float16* mid16  = (_Float16*)carve((size_t)NTOK * DFF * 2);       // gelu(ffn1)

    const dim3 blk256(256);
    auto g1d = [](size_t n) { return dim3((unsigned)((n + 255) / 256)); };

    // 1) weights fp32 -> f16
    cvt_f32_to_f16<<<g1d((size_t)DXR * DMODEL),   blk256, 0, stream>>>(w_inp, f_win,  DXR * DMODEL);
    cvt_f32_to_f16<<<g1d((size_t)DPROJ * DINNER), blk256, 0, stream>>>(w_xp,  f_wxp,  DPROJ * DINNER);
    cvt_f32_to_f16<<<g1d((size_t)DINNER * DTRANK),blk256, 0, stream>>>(w_dt,  f_wdt,  DINNER * DTRANK);
    cvt_f32_to_f16<<<g1d((size_t)DMODEL * DINNER),blk256, 0, stream>>>(w_out, f_wout, DMODEL * DINNER);
    cvt_f32_to_f16<<<g1d((size_t)DFF * DMODEL),   blk256, 0, stream>>>(w1,    f_w1,   DFF * DMODEL);
    cvt_f32_to_f16<<<g1d((size_t)DMODEL * DFF),   blk256, 0, stream>>>(w2,    f_w2,   DMODEL * DFF);

    const dim3 gblk(128);

    // 2) LN1 -> h1 (f16)
    ln_to_f16<<<dim3(NTOK), blk256, 0, stream>>>(src, n1g, n1b, h1_16, DMODEL);

    // 3) in_proj GEMM: (4096x768) x (3072x768)^T -> xres fp32
    gemm_f16_wmma<<<dim3(DXR / BN, NTOK / BM), gblk, 0, stream>>>(
        h1_16, f_win, NTOK, DXR, DMODEL, nullptr, nullptr, 0, xres, nullptr);

    // 4) depthwise conv + SiLU
    conv_silu<<<g1d((size_t)NTOK * DINNER), blk256, 0, stream>>>(xres, convw, convb, xs32, xs16);

    // 5) x_proj GEMM: (4096x1536) x (80x1536)^T -> projo fp32  (N=80 remainder path)
    gemm_f16_wmma<<<dim3((DPROJ + BN - 1) / BN, NTOK / BM), gblk, 0, stream>>>(
        xs16, f_wxp, NTOK, DPROJ, DINNER, nullptr, nullptr, 0, projo, nullptr);

    // 6) dt slice -> f16
    extract_dt<<<g1d((size_t)NTOK * DTRANK), blk256, 0, stream>>>(projo, dt16);

    // 7) dt_proj GEMM: (4096x48) x (1536x48)^T -> draw fp32  (K=48 remainder path)
    gemm_f16_wmma<<<dim3(DINNER / BN, NTOK / BM), gblk, 0, stream>>>(
        dt16, f_wdt, NTOK, DINNER, DTRANK, nullptr, nullptr, 0, draw, nullptr);

    // 8) selective scan (fp32 recurrence, 16 lanes per channel)
    selective_scan_kernel<<<dim3((NBATCH * DINNER) / 16), blk256, 0, stream>>>(
        draw, dt_b, A_log, D_p, xs32, projo, yssm);

    // 9) gate: y * silu(res) -> f16
    gate_silu<<<g1d((size_t)NTOK * DINNER), blk256, 0, stream>>>(yssm, xres, yg16);

    // 10) out_proj GEMM + residual: x1 = src + yg @ w_out^T
    gemm_f16_wmma<<<dim3(DMODEL / BN, NTOK / BM), gblk, 0, stream>>>(
        yg16, f_wout, NTOK, DMODEL, DINNER, nullptr, src, 0, x1, nullptr);

    // 11) LN2 -> h2 (f16)
    ln_to_f16<<<dim3(NTOK), blk256, 0, stream>>>(x1, n2g, n2b, h2_16, DMODEL);

    // 12) FFN1 GEMM + bias + exact GELU -> mid16 (f16 epilogue)
    gemm_f16_wmma<<<dim3(DFF / BN, NTOK / BM), gblk, 0, stream>>>(
        h2_16, f_w1, NTOK, DFF, DMODEL, b1, nullptr, 1, nullptr, mid16);

    // 13) FFN2 GEMM + bias + residual -> d_out (fp32)
    gemm_f16_wmma<<<dim3(DMODEL / BN, NTOK / BM), gblk, 0, stream>>>(
        mid16, f_w2, NTOK, DMODEL, DFF, b2, x1, 0, (float*)d_out, nullptr);
}
